// CircuitTrainingModel_43834436223238
// MI455X (gfx1250) — matrix operand order repaired and checked by reading, not compiled, not run on hardware
//
#include <hip/hip_runtime.h>

#define BB 4
#define NN 50000
#define NE 250000
#define HH 32
#define NL 3
#define GG 128
#define EPSF 1e-6f
#define NEGF -1e9f
#define ATT_CHUNKS 64

typedef __attribute__((ext_vector_type(16))) _Float16 v16h;
typedef __attribute__((ext_vector_type(8)))  _Float16 v8h;
typedef __attribute__((ext_vector_type(8)))  float    v8f;

// ---------------- utility ----------------
__global__ void k_zero(float* __restrict__ p, long n) {
    long i = (long)blockIdx.x * blockDim.x + threadIdx.x;
    if (i < n) p[i] = 0.f;
}

// ---------------- node feature embed: h_nodes = relu(nf @ wf + bf) ----------------
__global__ void __launch_bounds__(256)
k_embed(const float* __restrict__ nf, const float* __restrict__ wf,
        const float* __restrict__ bf, float* __restrict__ hn,
        _Float16* __restrict__ hn16) {
    __shared__ float sw[8 * 32];
    __shared__ float sb[32];
    int t = threadIdx.x;
    sw[t] = wf[t];
    if (t < 32) sb[t] = bf[t];
    __syncthreads();
    long n = (long)blockIdx.x * blockDim.x + t;
    if (n >= (long)BB * NN) return;
    float x[8];
#pragma unroll
    for (int c = 0; c < 8; c++) x[c] = nf[n * 8 + c];
#pragma unroll
    for (int h = 0; h < 32; h++) {
        float a = sb[h];
#pragma unroll
        for (int c = 0; c < 8; c++) a += x[c] * sw[c * 32 + h];
        a = fmaxf(a, 0.f);
        hn[n * 32 + h] = a;
        hn16[n * 32 + h] = (_Float16)a;
    }
}

// ---------------- h_meta = relu(metadata @ wm + bm) ----------------
__global__ void k_meta(const float* __restrict__ md, const float* __restrict__ wm,
                       const float* __restrict__ bm, float* __restrict__ hmeta) {
    int t = threadIdx.x;
    if (t >= BB * 32) return;
    int b = t >> 5, h = t & 31;
    float a = bm[h];
    for (int k = 0; k < 10; k++) a += md[b * 10 + k] * wm[k * 32 + h];
    hmeta[b * 32 + h] = fmaxf(a, 0.f);
}

// ---------------- degree counts (same every layer) ----------------
__global__ void k_counts(const int* __restrict__ ai, const int* __restrict__ aj,
                         float* __restrict__ counts) {
    long i = (long)blockIdx.x * blockDim.x + threadIdx.x;
    if (i >= (long)BB * NE) return;
    long b = i / NE;
    atomicAdd(&counts[b * NN + ai[i]], 1.f);
    atomicAdd(&counts[b * NN + aj[i]], 1.f);
}

// ---------------- pack edge weights into WMMA B layout (f16) ----------------
// wpack[l][ks][nt][lane][j] ; h=lane>>4: K = 32*ks + 8*h + (j&7) + 16*(j>>3), col N=(lane&15)+16*nt
//                             (h=0: K 0..7,16..23; h=1: K 8..15,24..31)
__global__ void k_packw(const float* __restrict__ ew, _Float16* __restrict__ wpack) {
    int i = blockIdx.x * blockDim.x + threadIdx.x;
    if (i >= NL * 2 * 2 * 32 * 16) return;
    int j    = i & 15;
    int lane = (i >> 4) & 31;
    int nt   = (i >> 9) & 1;
    int ks   = (i >> 10) & 1;
    int l    = i >> 11;
    int k = ks * 32 + ((lane >> 4) ? 8 : 0) + (j & 7) + ((j >> 3) * 16);
    int n = (lane & 15) + nt * 16;
    wpack[i] = (_Float16)ew[(l * 65 + k) * 32 + n];
}

// ---------------- fused edge MLP (WMMA) + scatter-add + last-layer mean ----------------
__global__ void __launch_bounds__(256)
k_edge_gcn(const _Float16* __restrict__ hn16,
           const int* __restrict__ ai, const int* __restrict__ aj,
           const float* __restrict__ aw,
           const _Float16* __restrict__ wpack,
           const float* __restrict__ ew, const float* __restrict__ eb,
           float* __restrict__ sums, float* __restrict__ mean_acc,
           int layer, int last) {
    __shared__ float he_tile[8][16][32];   // 16 KB
    int lane = threadIdx.x & 31;
    int wid  = threadIdx.x >> 5;
    int b    = blockIdx.y;
    long tileBase = ((long)blockIdx.x * 8 + wid) * 16;
    if (tileBase >= NE) return;            // wave-uniform exit, WMMA EXEC stays all-1s

    int er  = lane & 15;
    int grp = lane >> 4;
    long e  = (long)b * NE + tileBase + er;
    int ni = ai[e], nj = aj[e];
    float w = aw[e];
    _Float16 m = (w != 0.f) ? (_Float16)1.f : (_Float16)0.f;

    // A tiles: 16 edges x 32 halves; lane grp selects K sub-pattern {0..7,16..23}/{8..15,24..31}
    const _Float16* pi = hn16 + ((long)b * NN + ni) * 32 + grp * 8;
    const _Float16* pj = hn16 + ((long)b * NN + nj) * 32 + grp * 8;
    v8h e1lo = *(const v8h*)(pi);
    v8h e1hi = *(const v8h*)(pi + 16);
    v8h e2lo = *(const v8h*)(pj);
    v8h e2hi = *(const v8h*)(pj + 16);
    v16h a1, a2;
#pragma unroll
    for (int t = 0; t < 8; t++) {
        a1[t] = e1lo[t] * m; a1[t + 8] = e1hi[t] * m;
        a2[t] = e2lo[t] * m; a2[t + 8] = e2hi[t] * m;
    }
    const _Float16* wp = wpack + (long)layer * 2048 + lane * 16;
    v16h b00 = *(const v16h*)(wp);          // ks0 nt0
    v16h b01 = *(const v16h*)(wp + 512);    // ks0 nt1
    v16h b10 = *(const v16h*)(wp + 1024);   // ks1 nt0
    v16h b11 = *(const v16h*)(wp + 1536);   // ks1 nt1

    v8f c12a = {}, c12b = {}, c21a = {}, c21b = {};
    // f12 = [e1|e2] @ W ; f21 = [e2|e1] @ W   (K=64 as 2x K32 steps, 2 N-tiles)
    c12a = __builtin_amdgcn_wmma_f32_16x16x32_f16(false, a1, false, b00, (short)0, c12a, false, false);
    c12a = __builtin_amdgcn_wmma_f32_16x16x32_f16(false, a2, false, b10, (short)0, c12a, false, false);
    c12b = __builtin_amdgcn_wmma_f32_16x16x32_f16(false, a1, false, b01, (short)0, c12b, false, false);
    c12b = __builtin_amdgcn_wmma_f32_16x16x32_f16(false, a2, false, b11, (short)0, c12b, false, false);
    c21a = __builtin_amdgcn_wmma_f32_16x16x32_f16(false, a2, false, b00, (short)0, c21a, false, false);
    c21a = __builtin_amdgcn_wmma_f32_16x16x32_f16(false, a1, false, b10, (short)0, c21a, false, false);
    c21b = __builtin_amdgcn_wmma_f32_16x16x32_f16(false, a2, false, b01, (short)0, c21b, false, false);
    c21b = __builtin_amdgcn_wmma_f32_16x16x32_f16(false, a1, false, b11, (short)0, c21b, false, false);

    // epilogue: +w*W[64,:] + bias, relu, 0.5*(f12+f21)
    int n0 = er, n1 = er + 16;
    float bias0 = eb[layer * 32 + n0], bias1 = eb[layer * 32 + n1];
    float w64_0 = ew[(layer * 65 + 64) * 32 + n0];
    float w64_1 = ew[(layer * 65 + 64) * 32 + n1];
    float msum0 = 0.f, msum1 = 0.f;
#pragma unroll
    for (int r = 0; r < 8; r++) {
        int M = r + grp * 8;
        float wr = __shfl(w, M, 32);       // row M's edge weight
        float f12 = fmaxf(c12a[r] + wr * w64_0 + bias0, 0.f);
        float f21 = fmaxf(c21a[r] + wr * w64_0 + bias0, 0.f);
        float he0 = 0.5f * (f12 + f21);
        float g12 = fmaxf(c12b[r] + wr * w64_1 + bias1, 0.f);
        float g21 = fmaxf(c21b[r] + wr * w64_1 + bias1, 0.f);
        float he1 = 0.5f * (g12 + g21);
        he_tile[wid][M][n0] = he0;
        he_tile[wid][M][n1] = he1;
        msum0 += he0; msum1 += he1;
    }
    if (last) {                            // mean over edges (per column)
        msum0 += __shfl_xor(msum0, 16, 32);
        msum1 += __shfl_xor(msum1, 16, 32);
        if (grp == 0) {
            atomicAdd(&mean_acc[b * 32 + n0], msum0);
            atomicAdd(&mean_acc[b * 32 + n1], msum1);
        }
    }
    asm volatile("s_wait_dscnt 0x0" ::: "memory");  // in-wave LDS exchange
    // lane handles its own edge er, columns [grp*16, grp*16+16)
    float* si = sums + ((long)b * NN + ni) * 32;
    float* sj = sums + ((long)b * NN + nj) * 32;
    int c0 = grp * 16;
#pragma unroll
    for (int c = 0; c < 16; c++) {
        float hv = he_tile[wid][er][c0 + c];
        atomicAdd(si + c0 + c, hv);
        atomicAdd(sj + c0 + c, hv);
    }
}

// ---------------- h_nodes = sums / (counts + eps) ----------------
__global__ void k_finalize(const float* __restrict__ sums, const float* __restrict__ counts,
                           float* __restrict__ hn, _Float16* __restrict__ hn16) {
    long i = (long)blockIdx.x * blockDim.x + threadIdx.x;
    if (i >= (long)BB * NN * 32) return;
    float v = sums[i] / (counts[i >> 5] + EPSF);
    hn[i] = v;
    hn16[i] = (_Float16)v;
}

// ---------------- q = relu-free (hcur @ wq + bq); fold wk: qk = q·Wk^T, qb = q·bk ----------------
__global__ void k_query(const float* __restrict__ hn, const int* __restrict__ cur,
                        const float* __restrict__ wq, const float* __restrict__ bq,
                        const float* __restrict__ wk, const float* __restrict__ bk,
                        float* __restrict__ hcur, float* __restrict__ qkv,
                        float* __restrict__ qb) {
    __shared__ float sq[BB][32];
    __shared__ float sc[BB][32];
    int t = threadIdx.x;
    if (t >= BB * 32) return;
    int b = t >> 5, h = t & 31;
    float xc = hn[((long)b * NN + cur[b]) * 32 + h];
    sc[b][h] = xc;
    hcur[b * 32 + h] = xc;
    __syncthreads();
    float q = bq[h];
    for (int k = 0; k < 32; k++) q += sc[b][k] * wq[k * 32 + h];
    sq[b][h] = q;
    __syncthreads();
    float a = 0.f;                          // h plays role of channel c here
    for (int k = 0; k < 32; k++) a += sq[b][k] * wk[h * 32 + k];
    qkv[b * 32 + h] = a;
    if (h == 0) {
        float s = 0.f;
        for (int k = 0; k < 32; k++) s += sq[b][k] * bk[k];
        qb[b] = s;
    }
}

// ---------------- attention partials: online softmax, accumulate p-weighted x ----------------
__global__ void __launch_bounds__(256)
k_att_partial(const float* __restrict__ hn, const float* __restrict__ qk,
              const float* __restrict__ qb, float* __restrict__ part) {
    int b = blockIdx.y, ch = blockIdx.x;
    int t = threadIdx.x;
    __shared__ float sqk[32];
    __shared__ float sxa[32];
    __shared__ float red[256];
    if (t < 32) { sqk[t] = qk[b * 32 + t]; sxa[t] = 0.f; }
    __syncthreads();
    float qbb = qb[b];
    int per = (NN + ATT_CHUNKS - 1) / ATT_CHUNKS;
    int n0 = ch * per;
    int n1 = n0 + per; if (n1 > NN) n1 = NN;
    float m = -3.0e38f, s = 0.f;
    float xacc[32];
#pragma unroll
    for (int c = 0; c < 32; c++) xacc[c] = 0.f;
    for (int n = n0 + t; n < n1; n += 256) {
        const float* x = hn + ((long)b * NN + n) * 32;
        float scv = qbb;
        float xv[32];
#pragma unroll
        for (int c = 0; c < 32; c++) { xv[c] = x[c]; scv += xv[c] * sqk[c]; }
        if (scv > m) {
            float f = __expf(m - scv);
            s = s * f + 1.f;
#pragma unroll
            for (int c = 0; c < 32; c++) xacc[c] = xacc[c] * f + xv[c];
            m = scv;
        } else {
            float p = __expf(scv - m);
            s += p;
#pragma unroll
            for (int c = 0; c < 32; c++) xacc[c] += p * xv[c];
        }
    }
    red[t] = m; __syncthreads();
    for (int o = 128; o > 0; o >>= 1) { if (t < o) red[t] = fmaxf(red[t], red[t + o]); __syncthreads(); }
    float M = red[0]; __syncthreads();
    float f = (m > -1.0e38f) ? __expf(m - M) : 0.f;
    red[t] = s * f; __syncthreads();
    for (int o = 128; o > 0; o >>= 1) { if (t < o) red[t] += red[t + o]; __syncthreads(); }
    float S = red[0];
#pragma unroll
    for (int c = 0; c < 32; c++) atomicAdd(&sxa[c], xacc[c] * f);
    __syncthreads();
    float* pr = part + (long)(b * ATT_CHUNKS + ch) * 34;
    if (t == 0) { pr[0] = M; pr[1] = S; }
    if (t < 32) pr[2 + t] = sxa[t];
}

// ---------------- merge partials, apply Wv: h_att = (Σp x)/(Σp) @ Wv + bv ----------------
__global__ void k_att_merge(const float* __restrict__ part, const float* __restrict__ wvv,
                            const float* __restrict__ bv, float* __restrict__ hatt) {
    int b = blockIdx.x;
    int t = threadIdx.x;   // 64 threads == ATT_CHUNKS
    __shared__ float fm[ATT_CHUNKS], red[ATT_CHUNKS], xbar[32];
    float m = part[(long)(b * ATT_CHUNKS + t) * 34];
    red[t] = m; __syncthreads();
    for (int o = 32; o > 0; o >>= 1) { if (t < o) red[t] = fmaxf(red[t], red[t + o]); __syncthreads(); }
    float M = red[0]; __syncthreads();
    float fac = __expf(m - M);
    fm[t] = fac;
    red[t] = part[(long)(b * ATT_CHUNKS + t) * 34 + 1] * fac; __syncthreads();
    for (int o = 32; o > 0; o >>= 1) { if (t < o) red[t] += red[t + o]; __syncthreads(); }
    float S = red[0]; __syncthreads();
    if (t < 32) {
        float a = 0.f;
        for (int c = 0; c < ATT_CHUNKS; c++) a += part[(long)(b * ATT_CHUNKS + c) * 34 + 2 + t] * fm[c];
        xbar[t] = a / S;
    }
    __syncthreads();
    if (t < 32) {
        float a = bv[t];
        for (int c = 0; c < 32; c++) a += xbar[c] * wvv[c * 32 + t];
        hatt[b * 32 + t] = a;
    }
}

// ---------------- value head + policy FC ----------------
__global__ void __launch_bounds__(256)
k_heads(const float* __restrict__ hmeta, const float* __restrict__ mean_acc,
        const float* __restrict__ hcur, const float* __restrict__ hatt,
        const float* __restrict__ vw1, const float* __restrict__ vb1,
        const float* __restrict__ vw2, const float* __restrict__ vb2,
        const float* __restrict__ vw3, const float* __restrict__ vb3,
        const float* __restrict__ pw1, const float* __restrict__ pb1,
        float* __restrict__ value_out, float* __restrict__ x0) {
    int b = blockIdx.x;
    int t = threadIdx.x;
    __shared__ float h[128];
    __shared__ float v1[32], v2[8];
    if (t < 32)       h[t] = hmeta[b * 32 + t];
    else if (t < 64)  h[t] = mean_acc[b * 32 + (t - 32)] * (1.f / NE);
    else if (t < 96)  h[t] = hcur[b * 32 + (t - 64)];
    else if (t < 128) h[t] = hatt[b * 32 + (t - 96)];
    __syncthreads();
    if (t < 32) {
        float a = vb1[t];
        for (int k = 0; k < 128; k++) a += h[k] * vw1[k * 32 + t];
        v1[t] = fmaxf(a, 0.f);
    }
    __syncthreads();
    if (t < 8) {
        float a = vb2[t];
        for (int k = 0; k < 32; k++) a += v1[k] * vw2[k * 8 + t];
        v2[t] = fmaxf(a, 0.f);
    }
    __syncthreads();
    if (t == 0) {
        float a = vb3[0];
        for (int k = 0; k < 8; k++) a += v2[k] * vw3[k];
        value_out[b] = a;
    }
    for (int o = t; o < 2048; o += 256) {
        float a = pb1[o];
        for (int k = 0; k < 128; k++) a += h[k] * pw1[k * 2048 + o];
        x0[(long)b * 2048 + o] = fmaxf(a, 0.f);
    }
}

// ---------------- transposed conv (lhs-dilated conv), relu, optional final mask ----------------
__global__ void k_deconv(const float* __restrict__ x, const float* __restrict__ kw,
                         const float* __restrict__ kb, float* __restrict__ y,
                         int Hin, int CI, int CO, int stride, int padA,
                         const int* __restrict__ mask, float* __restrict__ logits) {
    int Hout = Hin * stride;
    long total = (long)BB * Hout * Hout * CO;
    long i = (long)blockIdx.x * blockDim.x + threadIdx.x;
    if (i >= total) return;
    int co = i % CO;
    long r = i / CO;
    int ow = r % Hout; r /= Hout;
    int oh = r % Hout; r /= Hout;
    int b = (int)r;
    float a = kb[co];
    for (int kh = 0; kh < 3; kh++) {
        int dh = oh + kh - padA;
        if (dh < 0 || (dh % stride) || dh / stride >= Hin) continue;
        int ih = dh / stride;
        for (int kwt = 0; kwt < 3; kwt++) {
            int dw = ow + kwt - padA;
            if (dw < 0 || (dw % stride) || dw / stride >= Hin) continue;
            int iw = dw / stride;
            const float* xp  = x + (((long)b * Hin + ih) * Hin + iw) * CI;
            const float* wp2 = kw + (long)(kh * 3 + kwt) * CI * CO + co;
            for (int ci = 0; ci < CI; ci++) a += xp[ci] * wp2[ci * CO];
        }
    }
    a = fmaxf(a, 0.f);
    if (logits) {
        long p = (long)oh * GG + ow;  // CO == 1
        logits[(long)b * GG * GG + p] = mask[(long)b * GG * GG + p] ? a : NEGF;
    } else {
        y[i] = a;
    }
}

// ---------------- launch ----------------
extern "C" void kernel_launch(void* const* d_in, const int* in_sizes, int n_in,
                              void* d_out, int out_size, void* d_ws, size_t ws_size,
                              hipStream_t stream) {
    const float* node_features = (const float*)d_in[0];
    const float* metadata      = (const float*)d_in[1];
    const int*   adj_i         = (const int*)d_in[2];
    const int*   adj_j         = (const int*)d_in[3];
    const float* adj_w         = (const float*)d_in[4];
    const int*   cur           = (const int*)d_in[5];
    const int*   mask          = (const int*)d_in[6];
    const float* wm  = (const float*)d_in[7];
    const float* bm  = (const float*)d_in[8];
    const float* wf  = (const float*)d_in[9];
    const float* bf  = (const float*)d_in[10];
    const float* edge_w = (const float*)d_in[11];
    const float* edge_b = (const float*)d_in[12];
    const float* wq = (const float*)d_in[13];
    const float* bq = (const float*)d_in[14];
    const float* wk = (const float*)d_in[15];
    const float* bk = (const float*)d_in[16];
    const float* wv = (const float*)d_in[17];
    const float* bv = (const float*)d_in[18];
    const float* vw1 = (const float*)d_in[19];
    const float* vb1 = (const float*)d_in[20];
    const float* vw2 = (const float*)d_in[21];
    const float* vb2 = (const float*)d_in[22];
    const float* vw3 = (const float*)d_in[23];
    const float* vb3 = (const float*)d_in[24];
    const float* pw1 = (const float*)d_in[25];
    const float* pb1 = (const float*)d_in[26];
    const float* dk[5] = {(const float*)d_in[27], (const float*)d_in[29], (const float*)d_in[31],
                          (const float*)d_in[33], (const float*)d_in[35]};
    const float* db[5] = {(const float*)d_in[28], (const float*)d_in[30], (const float*)d_in[32],
                          (const float*)d_in[34], (const float*)d_in[36]};

    char* base = (char*)d_ws;
    size_t off = 0;
    auto alloc = [&](size_t bytes) -> char* {
        size_t o = (off + 255) & ~(size_t)255;
        off = o + bytes;
        return base + o;
    };
    float*    hn       = (float*)alloc((size_t)BB * NN * 32 * 4);
    _Float16* hn16     = (_Float16*)alloc((size_t)BB * NN * 32 * 2);
    float*    sums     = (float*)alloc((size_t)BB * NN * 32 * 4);
    float*    counts   = (float*)alloc((size_t)BB * NN * 4);
    _Float16* wpack    = (_Float16*)alloc((size_t)NL * 2 * 2 * 512 * 2);
    float*    hmeta    = (float*)alloc(BB * 32 * 4);
    float*    hcur     = (float*)alloc(BB * 32 * 4);
    float*    qkbuf    = (float*)alloc(BB * 32 * 4);
    float*    qbbuf    = (float*)alloc(BB * 4);
    float*    mean_acc = (float*)alloc(BB * 32 * 4);
    float*    hatt     = (float*)alloc(BB * 32 * 4);
    float*    part     = (float*)alloc((size_t)BB * ATT_CHUNKS * 34 * 4);
    float*    x0       = (float*)alloc((size_t)BB * 8 * 8 * 32 * 4);
    float*    x1       = (float*)alloc((size_t)BB * 16 * 16 * 16 * 4);
    float*    x2       = (float*)alloc((size_t)BB * 32 * 32 * 8 * 4);
    float*    x3       = (float*)alloc((size_t)BB * 64 * 64 * 4 * 4);
    float*    x4       = (float*)alloc((size_t)BB * 128 * 128 * 2 * 4);

    float* logits    = (float*)d_out;
    float* value_out = logits + (size_t)BB * GG * GG;

    // init / preprocessing
    k_zero<<<(BB * NN + 255) / 256, 256, 0, stream>>>(counts, (long)BB * NN);
    k_zero<<<1, 256, 0, stream>>>(mean_acc, BB * 32);
    k_counts<<<((long)BB * NE + 255) / 256, 256, 0, stream>>>(adj_i, adj_j, counts);
    k_packw<<<(NL * 2 * 2 * 512 + 255) / 256, 256, 0, stream>>>(edge_w, wpack);
    k_embed<<<(BB * NN + 255) / 256, 256, 0, stream>>>(node_features, wf, bf, hn, hn16);
    k_meta<<<1, 128, 0, stream>>>(metadata, wm, bm, hmeta);

    // GCN layers: WMMA edge MLP fused with scatter-add
    int gx = (NE + 127) / 128;  // 8 waves/block, 16 edges/wave
    for (int l = 0; l < NL; l++) {
        k_zero<<<((long)BB * NN * 32 + 255) / 256, 256, 0, stream>>>(sums, (long)BB * NN * 32);
        k_edge_gcn<<<dim3(gx, BB), 256, 0, stream>>>(hn16, adj_i, adj_j, adj_w, wpack,
                                                     edge_w, edge_b, sums, mean_acc,
                                                     l, (l == NL - 1) ? 1 : 0);
        k_finalize<<<((long)BB * NN * 32 + 255) / 256, 256, 0, stream>>>(sums, counts, hn, hn16);
    }

    // attention
    k_query<<<1, 128, 0, stream>>>(hn, cur, wq, bq, wk, bk, hcur, qkbuf, qbbuf);
    k_att_partial<<<dim3(ATT_CHUNKS, BB), 256, 0, stream>>>(hn, qkbuf, qbbuf, part);
    k_att_merge<<<BB, ATT_CHUNKS, 0, stream>>>(part, wv, bv, hatt);

    // heads
    k_heads<<<BB, 256, 0, stream>>>(hmeta, mean_acc, hcur, hatt, vw1, vb1, vw2, vb2,
                                    vw3, vb3, pw1, pb1, value_out, x0);

    // deconv stack
    long n1 = (long)BB * 16 * 16 * 16;
    k_deconv<<<(n1 + 255) / 256, 256, 0, stream>>>(x0, dk[0], db[0], x1, 8, 32, 16, 2, 2, nullptr, nullptr);
    long n2 = (long)BB * 32 * 32 * 8;
    k_deconv<<<(n2 + 255) / 256, 256, 0, stream>>>(x1, dk[1], db[1], x2, 16, 16, 8, 2, 2, nullptr, nullptr);
    long n3 = (long)BB * 64 * 64 * 4;
    k_deconv<<<(n3 + 255) / 256, 256, 0, stream>>>(x2, dk[2], db[2], x3, 32, 8, 4, 2, 2, nullptr, nullptr);
    long n4 = (long)BB * 128 * 128 * 2;
    k_deconv<<<(n4 + 255) / 256, 256, 0, stream>>>(x3, dk[3], db[3], x4, 64, 4, 2, 2, 2, nullptr, nullptr);
    long n5 = (long)BB * 128 * 128 * 1;
    k_deconv<<<(n5 + 255) / 256, 256, 0, stream>>>(x4, dk[4], db[4], nullptr, 128, 2, 1, 1, 1, mask, logits);
}
